// dark_light_46918222742395
// MI455X (gfx1250) — matrix operand order repaired and checked by reading, not compiled, not run on hardware
//
#include <hip/hip_runtime.h>
#include <hip/hip_bf16.h>

typedef __attribute__((ext_vector_type(16))) _Float16 v16h;
typedef __attribute__((ext_vector_type(8)))  _Float16 v8h;
typedef __attribute__((ext_vector_type(8)))  float    v8f;

#define D_MODEL 512
#define NLAT    128
#define INNER   1024
#define H_PA    8
#define DH_PA   128
#define BATCH   8
#define MTOK    4096
#define JLEN    (MTOK + NLAT)   // 4224
#define ROWS_KV (BATCH * JLEN)  // 33792
#define ROWS_L  (BATCH * NLAT)  // 1024

// ---------------------------------------------------------------- utilities
__device__ __forceinline__ float wave_sum32(float v) {
#pragma unroll
  for (int m = 16; m >= 1; m >>= 1) v += __shfl_xor(v, m, 32);
  return v;
}
__device__ __forceinline__ float half_max16(float v) {
#pragma unroll
  for (int m = 8; m >= 1; m >>= 1) v = fmaxf(v, __shfl_xor(v, m, 32));
  return v;
}
__device__ __forceinline__ float half_sum16(float v) {
#pragma unroll
  for (int m = 8; m >= 1; m >>= 1) v += __shfl_xor(v, m, 32);
  return v;
}

// WMMA 16-bit A-frag K offset for slot s (0..15) given lane-half.
// s=0..7 -> k = s + half*8; s=8..15 -> k = 8 + s + half*8 (AMD table order)
__device__ __forceinline__ int a_koff(int s, int half_) {
  return (s & 7) + ((s >> 3) << 4) + (half_ << 3);
}

// ---------------------------------------------------------------- f32 -> f16 transpose
// src: f32 [N x K] row-major -> dst: f16 [K x N] row-major (coalesced writes, L2-cached reads)
__global__ void cvt_t_kernel(const float* __restrict__ src,
                             _Float16* __restrict__ dst, int N, int K) {
  int idx = blockIdx.x * 256 + threadIdx.x;
  if (idx >= N * K) return;
  int k = idx / N, n = idx - k * N;
  dst[idx] = (_Float16)src[(size_t)n * K + k];
}

// ---------------------------------------------------------------- LayerNorm (512 cols, wave per row)
__global__ void ln512_kernel(const float* __restrict__ in,
                             const float* __restrict__ g,
                             const float* __restrict__ bt,
                             _Float16* __restrict__ outh, int remap) {
  int row  = blockIdx.x * 8 + (threadIdx.x >> 5);
  int lane = threadIdx.x & 31;
  const float* x = in + (size_t)row * D_MODEL;
  float v[16]; float s = 0.f;
#pragma unroll
  for (int i = 0; i < 16; ++i) { v[i] = x[lane + 32 * i]; s += v[i]; }
  s = wave_sum32(s);
  float mean = s * (1.f / 512.f);
  float ss = 0.f;
#pragma unroll
  for (int i = 0; i < 16; ++i) { float d = v[i] - mean; ss += d * d; }
  ss = wave_sum32(ss);
  float rstd = rsqrtf(ss * (1.f / 512.f) + 1e-5f);
  size_t orow = remap ? ((size_t)(row >> 12) * JLEN + (size_t)(row & 4095))
                      : (size_t)row;
  _Float16* o = outh + orow * D_MODEL;
#pragma unroll
  for (int i = 0; i < 16; ++i) {
    int c = lane + 32 * i;
    o[c] = (_Float16)((v[i] - mean) * rstd * g[c] + bt[c]);
  }
}

// Double LN for latents: LN(latents_p, nl) then LN(., pa_nl)  -> f16 [128,512]
__global__ void ln512_double_kernel(const float* __restrict__ in,
                                    const float* __restrict__ g1,
                                    const float* __restrict__ b1,
                                    const float* __restrict__ g2,
                                    const float* __restrict__ b2,
                                    _Float16* __restrict__ outh) {
  int row  = blockIdx.x * 8 + (threadIdx.x >> 5);
  int lane = threadIdx.x & 31;
  const float* x = in + (size_t)row * D_MODEL;
  float v[16]; float s = 0.f;
#pragma unroll
  for (int i = 0; i < 16; ++i) { v[i] = x[lane + 32 * i]; s += v[i]; }
  s = wave_sum32(s);
  float mean = s * (1.f / 512.f);
  float ss = 0.f;
#pragma unroll
  for (int i = 0; i < 16; ++i) { float d = v[i] - mean; ss += d * d; }
  ss = wave_sum32(ss);
  float rstd = rsqrtf(ss * (1.f / 512.f) + 1e-5f);
  float y[16]; s = 0.f;
#pragma unroll
  for (int i = 0; i < 16; ++i) {
    int c = lane + 32 * i;
    y[i] = (v[i] - mean) * rstd * g1[c] + b1[c];
    s += y[i];
  }
  s = wave_sum32(s);
  mean = s * (1.f / 512.f);
  ss = 0.f;
#pragma unroll
  for (int i = 0; i < 16; ++i) { float d = y[i] - mean; ss += d * d; }
  ss = wave_sum32(ss);
  rstd = rsqrtf(ss * (1.f / 512.f) + 1e-5f);
  _Float16* o = outh + (size_t)row * D_MODEL;
#pragma unroll
  for (int i = 0; i < 16; ++i) {
    int c = lane + 32 * i;
    o[c] = (_Float16)((y[i] - mean) * rstd * g2[c] + b2[c]);
  }
}

__global__ void bcast_lat_kernel(const _Float16* __restrict__ l,
                                 _Float16* __restrict__ Akv) {
  int idx = blockIdx.x * 256 + threadIdx.x;           // 8*128*512 total
  int b   = idx / (NLAT * D_MODEL);
  int rem = idx - b * (NLAT * D_MODEL);
  Akv[((size_t)b * JLEN + MTOK) * D_MODEL + rem] = l[rem];
}

// ---------------------------------------------------------------- generic WMMA GEMM
// C[row,col] = act((sum_k A[row,k]*Wt[k,col] + bias[col]) * scale) + resid[row,col]
// A:  f16 [rows x K] row-major.  Wt: f16 [K x N] row-major (pre-transposed weights).
// Tiles 64x64, BK=64 (two 32-K chunks per barrier pair). 128 threads = 4 waves.
// mode 0: normal outputs (outH f16 and/or outF f32).
// mode 2: kv-split: col<1024 -> outH = k_t[col*ROWS_KV + row], else outH2 = v[row*1024 + col-1024]
__global__ void gemm64_kernel(const _Float16* __restrict__ A,
                              const _Float16* __restrict__ Wt,
                              const float* __restrict__ bias,
                              const float* __restrict__ resid,
                              float scale, int act,
                              _Float16* __restrict__ outH,
                              float* __restrict__ outF,
                              _Float16* __restrict__ outH2,
                              int mode, int N, int K) {
  __shared__ _Float16 As[64 * 72];   // [m 0..63][k 0..63], stride 72 (144B rows)
  __shared__ _Float16 Ws[64 * 72];   // [k 0..63][n 0..63], stride 72
  const int rowBase = blockIdx.x * 64;
  const int colBase = blockIdx.y * 64;
  const int w     = threadIdx.x >> 5;
  const int lane  = threadIdx.x & 31;
  const int lm    = lane & 15;
  const int half_ = lane >> 4;
  const int kl    = lm + half_ * 16;   // 0..31 within a 32-K chunk

  v8f acc[4];
#pragma unroll
  for (int nt = 0; nt < 4; ++nt)
#pragma unroll
    for (int r = 0; r < 8; ++r) acc[nt][r] = 0.f;

  const int ldRow = threadIdx.x >> 1;        // 0..63
  const int ldC0  = (threadIdx.x & 1) * 32;  // 0 or 32

  for (int k0 = 0; k0 < K; k0 += 64) {
    // cooperative tile loads: 4x v8h per thread per tile, all 16B-aligned
    const v8h* sa = (const v8h*)(A + (size_t)(rowBase + ldRow) * K + k0 + ldC0);
    v8h a0 = sa[0], a1 = sa[1], a2 = sa[2], a3 = sa[3];
    const v8h* sw = (const v8h*)(Wt + (size_t)(k0 + ldRow) * N + colBase + ldC0);
    v8h w0 = sw[0], w1 = sw[1], w2 = sw[2], w3 = sw[3];
    // prefetch next K-slab into cache while this slab computes
    if (k0 + 64 < K) {
      __builtin_prefetch(A + (size_t)(rowBase + ldRow) * K + k0 + 64 + ldC0, 0, 0);
      __builtin_prefetch(Wt + (size_t)(k0 + 64 + ldRow) * N + colBase + ldC0, 0, 0);
    }
    *(v8h*)&As[ldRow * 72 + ldC0]      = a0;
    *(v8h*)&As[ldRow * 72 + ldC0 + 8]  = a1;
    *(v8h*)&As[ldRow * 72 + ldC0 + 16] = a2;
    *(v8h*)&As[ldRow * 72 + ldC0 + 24] = a3;
    *(v8h*)&Ws[ldRow * 72 + ldC0]      = w0;
    *(v8h*)&Ws[ldRow * 72 + ldC0 + 8]  = w1;
    *(v8h*)&Ws[ldRow * 72 + ldC0 + 16] = w2;
    *(v8h*)&Ws[ldRow * 72 + ldC0 + 24] = w3;
    __syncthreads();

#pragma unroll
    for (int kc = 0; kc < 2; ++kc) {
      // load A-frag + all 4 B-frags first, then an uninterrupted WMMA chain
      v16h a;
#pragma unroll
      for (int s = 0; s < 16; ++s)
        a[s] = As[(w * 16 + lm) * 72 + kc * 32 + a_koff(s, half_)];
      v16h b0, b1, b2, b3;
#pragma unroll
      for (int s = 0; s < 16; ++s) {
        b0[s] = Ws[(kc * 32 + a_koff(s, half_)) * 72 +  0 + lm];
        b1[s] = Ws[(kc * 32 + a_koff(s, half_)) * 72 + 16 + lm];
        b2[s] = Ws[(kc * 32 + a_koff(s, half_)) * 72 + 32 + lm];
        b3[s] = Ws[(kc * 32 + a_koff(s, half_)) * 72 + 48 + lm];
      }
      acc[0] = __builtin_amdgcn_wmma_f32_16x16x32_f16(false, a, false, b0,
                                                      (short)0, acc[0], false, false);
      acc[1] = __builtin_amdgcn_wmma_f32_16x16x32_f16(false, a, false, b1,
                                                      (short)0, acc[1], false, false);
      acc[2] = __builtin_amdgcn_wmma_f32_16x16x32_f16(false, a, false, b2,
                                                      (short)0, acc[2], false, false);
      acc[3] = __builtin_amdgcn_wmma_f32_16x16x32_f16(false, a, false, b3,
                                                      (short)0, acc[3], false, false);
    }
    __syncthreads();
  }

  // epilogue
#pragma unroll
  for (int nt = 0; nt < 4; ++nt) {
#pragma unroll
    for (int r = 0; r < 8; ++r) {
      int row = rowBase + w * 16 + half_ * 8 + r;
      int col = colBase + nt * 16 + lm;
      float v = acc[nt][r];
      if (bias)  v += bias[col];
      if (mode == 2) {
        if (col < INNER) outH [(size_t)col * ROWS_KV + row] = (_Float16)v;  // k^T
        else             outH2[(size_t)row * INNER + (col - INNER)] = (_Float16)v;  // v
      } else {
        v *= scale;
        if (act == 1) v = v / (1.f + __expf(-1.702f * v));   // qGELU
        if (resid) v += resid[(size_t)row * N + col];
        if (outF) outF[(size_t)row * N + col] = v;
        if (outH) outH[(size_t)row * N + col] = (_Float16)v;
      }
    }
  }
}

// ---------------------------------------------------------------- flash perceiver attention
// grid: (B*H, NLAT/16), block: 1 wave.
// q:  f16 [128,1024] (dh^-0.5 prefolded)
// kT: f16 [1024 dh][ROWS_KV]  (k transposed: row = b*4224 + j)
// vH: f16 [ROWS_KV][1024]
// o:  f16 [B,128,1024]
__global__ void attn_kernel(const _Float16* __restrict__ q,
                            const _Float16* __restrict__ kT,
                            const _Float16* __restrict__ vH,
                            _Float16* __restrict__ o) {
  __shared__ _Float16 pT[16 * 40];   // stride 40 -> 16B aligned rows, conflict-free
  const int b  = blockIdx.x >> 3;
  const int h  = blockIdx.x & 7;
  const int n0 = blockIdx.y * 16;
  const int lane  = threadIdx.x & 31;
  const int lm    = lane & 15;
  const int half_ = lane >> 4;
  const int kkl   = lm + half_ * 16;   // 0..31

  // q A-frags over dh=128 (4 K-chunks of 32), held in registers
  v16h aq[4];
#pragma unroll
  for (int kc = 0; kc < 4; ++kc)
#pragma unroll
    for (int s = 0; s < 16; ++s)
      aq[kc][s] = q[(size_t)(n0 + lm) * INNER + h * DH_PA + kc * 32 + a_koff(s, half_)];

  float mrun[8], lsum[8];
  v8f acc_o[8];
#pragma unroll
  for (int r = 0; r < 8; ++r) { mrun[r] = -1e30f; lsum[r] = 0.f; }
#pragma unroll
  for (int c = 0; c < 8; ++c)
#pragma unroll
    for (int r = 0; r < 8; ++r) acc_o[c][r] = 0.f;

  const size_t jbase = (size_t)b * JLEN;

  for (int j0 = 0; j0 < JLEN; j0 += 32) {
    // ---- scores S[16n x 32j];  k B-frag: lane = j, slots = dh (a_koff order)
    v8f s0, s1;
#pragma unroll
    for (int r = 0; r < 8; ++r) { s0[r] = 0.f; s1[r] = 0.f; }
#pragma unroll
    for (int kp2 = 0; kp2 < 2; ++kp2) {   // two kc at a time: 4 frags then 4 wmma
      const _Float16* kpa = kT + (size_t)(h * DH_PA + (2 * kp2 + 0) * 32) * ROWS_KV + jbase + j0;
      const _Float16* kpb = kT + (size_t)(h * DH_PA + (2 * kp2 + 1) * 32) * ROWS_KV + jbase + j0;
      v16h bk0a, bk1a, bk0b, bk1b;
#pragma unroll
      for (int s = 0; s < 16; ++s) {
        bk0a[s] = kpa[(size_t)a_koff(s, half_) * ROWS_KV + lm]; bk1a[s] = kpa[(size_t)a_koff(s, half_) * ROWS_KV + 16 + lm];
        bk0b[s] = kpb[(size_t)a_koff(s, half_) * ROWS_KV + lm]; bk1b[s] = kpb[(size_t)a_koff(s, half_) * ROWS_KV + 16 + lm];
      }
      s0 = __builtin_amdgcn_wmma_f32_16x16x32_f16(false, aq[2 * kp2 + 0], false, bk0a,
                                                  (short)0, s0, false, false);
      s1 = __builtin_amdgcn_wmma_f32_16x16x32_f16(false, aq[2 * kp2 + 0], false, bk1a,
                                                  (short)0, s1, false, false);
      s0 = __builtin_amdgcn_wmma_f32_16x16x32_f16(false, aq[2 * kp2 + 1], false, bk0b,
                                                  (short)0, s0, false, false);
      s1 = __builtin_amdgcn_wmma_f32_16x16x32_f16(false, aq[2 * kp2 + 1], false, bk1b,
                                                  (short)0, s1, false, false);
    }

    // ---- online softmax (rows live in 16-lane half groups, slot = row)
    float p0[8], p1[8], corr[8];
#pragma unroll
    for (int r = 0; r < 8; ++r) {
      float rm = half_max16(fmaxf(s0[r], s1[r]));
      float mn = fmaxf(mrun[r], rm);
      corr[r]  = __expf(mrun[r] - mn);
      mrun[r]  = mn;
      p0[r] = __expf(s0[r] - mn);
      p1[r] = __expf(s1[r] - mn);
      lsum[r] = lsum[r] * corr[r] + half_sum16(p0[r] + p1[r]);
    }
#pragma unroll
    for (int c = 0; c < 8; ++c)
#pragma unroll
      for (int r = 0; r < 8; ++r) acc_o[c][r] *= corr[r];

    // ---- P (C-layout) -> A-layout via LDS (DS in-order within a wave)
#pragma unroll
    for (int r = 0; r < 8; ++r) {
      pT[(half_ * 8 + r) * 40 + lm]      = (_Float16)p0[r];
      pT[(half_ * 8 + r) * 40 + 16 + lm] = (_Float16)p1[r];
    }
    v16h ap;
#pragma unroll
    for (int s = 0; s < 16; ++s) ap[s] = pT[lm * 40 + a_koff(s, half_)];

    // ---- O += P @ V   (V B-frag: lane = dh, slots = j (a_koff order))
#pragma unroll
    for (int cp = 0; cp < 2; ++cp) {      // four frags then four wmma
      v16h bv[4];
#pragma unroll
      for (int c = 0; c < 4; ++c) {
        const _Float16* vp =
            vH + (jbase + j0) * INNER + h * DH_PA + (cp * 4 + c) * 16 + lm;
#pragma unroll
        for (int s = 0; s < 16; ++s) bv[c][s] = vp[(size_t)a_koff(s, half_) * INNER];
      }
#pragma unroll
      for (int c = 0; c < 4; ++c)
        acc_o[cp * 4 + c] = __builtin_amdgcn_wmma_f32_16x16x32_f16(
            false, ap, false, bv[c], (short)0, acc_o[cp * 4 + c], false, false);
    }
  }

  // ---- write O / lsum
#pragma unroll
  for (int c = 0; c < 8; ++c)
#pragma unroll
    for (int r = 0; r < 8; ++r) {
      int n = n0 + half_ * 8 + r;
      o[((size_t)b * NLAT + n) * INNER + h * DH_PA + c * 16 + lm] =
          (_Float16)(acc_o[c][r] / lsum[r]);
    }
}

// ---------------------------------------------------------------- batch-axis MHA (the torch bug)
// grid 128 (n), block 256: thread = (h, i, quad). Each quad handles 16 of 64 dims;
// score partials reduced within the quad via shfl_xor (same wave).
__global__ void mha_batch_kernel(const float* __restrict__ qkv,
                                 _Float16* __restrict__ o2h) {
  int n    = blockIdx.x;
  int t    = threadIdx.x;
  int quad = t & 3;            // d-quarter
  int i    = (t >> 2) & 7;     // query batch index
  int h    = t >> 5;           // head
  const float scale = 0.125f;  // 64^-0.5
  const int dOff = h * 64 + quad * 16;

  float qv[16];
  const float* qp = qkv + ((size_t)i * NLAT + n) * 1536 + dOff;
#pragma unroll
  for (int d = 0; d < 16; ++d) qv[d] = qp[d] * scale;

  float sc[8];
#pragma unroll
  for (int j = 0; j < 8; ++j) {
    const float* kp = qkv + ((size_t)j * NLAT + n) * 1536 + 512 + dOff;
    float s = 0.f;
#pragma unroll
    for (int d = 0; d < 16; ++d) s += qv[d] * kp[d];
    s += __shfl_xor(s, 1, 32);
    s += __shfl_xor(s, 2, 32);
    sc[j] = s;
  }
  float mx = sc[0];
#pragma unroll
  for (int j = 1; j < 8; ++j) mx = fmaxf(mx, sc[j]);
  float sum = 0.f;
#pragma unroll
  for (int j = 0; j < 8; ++j) { sc[j] = __expf(sc[j] - mx); sum += sc[j]; }
  float inv = 1.f / sum;

  float ov[16];
#pragma unroll
  for (int d = 0; d < 16; ++d) ov[d] = 0.f;
#pragma unroll
  for (int j = 0; j < 8; ++j) {
    const float* vp = qkv + ((size_t)j * NLAT + n) * 1536 + 1024 + dOff;
    float p = sc[j] * inv;
#pragma unroll
    for (int d = 0; d < 16; ++d) ov[d] += p * vp[d];
  }
  _Float16* op = o2h + ((size_t)i * NLAT + n) * 512 + dOff;
#pragma unroll
  for (int d = 0; d < 16; ++d) op[d] = (_Float16)ov[d];
}

// ---------------------------------------------------------------- launch
extern "C" void kernel_launch(void* const* d_in, const int* in_sizes, int n_in,
                              void* d_out, int out_size, void* d_ws, size_t ws_size,
                              hipStream_t stream) {
  const float* visual = (const float*)d_in[0];
  const float* lat_p  = (const float*)d_in[1];
  const float* nl_g   = (const float*)d_in[2];
  const float* nl_b   = (const float*)d_in[3];
  const float* nm_g   = (const float*)d_in[4];
  const float* nm_b   = (const float*)d_in[5];
  const float* pl_g   = (const float*)d_in[6];
  const float* pl_b   = (const float*)d_in[7];
  const float* Wq     = (const float*)d_in[8];
  const float* bq     = (const float*)d_in[9];
  const float* Wkv    = (const float*)d_in[10];
  const float* bkv    = (const float*)d_in[11];
  const float* Wo     = (const float*)d_in[12];
  const float* bo     = (const float*)d_in[13];
  const float* Wip    = (const float*)d_in[14];
  const float* bip    = (const float*)d_in[15];
  const float* Wop    = (const float*)d_in[16];
  const float* bop    = (const float*)d_in[17];
  const float* ln1_g  = (const float*)d_in[18];
  const float* ln1_b  = (const float*)d_in[19];
  const float* ln2_g  = (const float*)d_in[20];
  const float* ln2_b  = (const float*)d_in[21];
  const float* Wfc    = (const float*)d_in[22];
  const float* bfc    = (const float*)d_in[23];
  const float* Wpr    = (const float*)d_in[24];
  const float* bpr    = (const float*)d_in[25];

  char* wp = (char*)d_ws;
  auto alloc = [&](size_t bytes) -> void* {
    void* r = (void*)wp;
    wp += (bytes + 255) & ~(size_t)255;
    return r;
  };
  // pre-transposed f16 weights: Wt [K x N]
  _Float16* WqT  = (_Float16*)alloc((size_t)D_MODEL * INNER * 2);
  _Float16* WkvT = (_Float16*)alloc((size_t)D_MODEL * 2 * INNER * 2);
  _Float16* WoT  = (_Float16*)alloc((size_t)INNER * D_MODEL * 2);
  _Float16* WipT = (_Float16*)alloc((size_t)D_MODEL * 3 * D_MODEL * 2);
  _Float16* WopT = (_Float16*)alloc((size_t)D_MODEL * D_MODEL * 2);
  _Float16* WfcT = (_Float16*)alloc((size_t)D_MODEL * 4 * D_MODEL * 2);
  _Float16* WprT = (_Float16*)alloc((size_t)4 * D_MODEL * D_MODEL * 2);
  _Float16* l_h  = (_Float16*)alloc((size_t)NLAT * D_MODEL * 2);
  _Float16* Akv  = (_Float16*)alloc((size_t)ROWS_KV * D_MODEL * 2);
  _Float16* kT   = (_Float16*)alloc((size_t)INNER * ROWS_KV * 2);
  _Float16* vH   = (_Float16*)alloc((size_t)ROWS_KV * INNER * 2);
  _Float16* q_h  = (_Float16*)alloc((size_t)NLAT * INNER * 2);
  _Float16* o_h  = (_Float16*)alloc((size_t)ROWS_L * INNER * 2);
  float*    lat  = (float*)   alloc((size_t)ROWS_L * D_MODEL * 4);
  _Float16* y_h  = (_Float16*)alloc((size_t)ROWS_L * D_MODEL * 2);
  float*    qkv  = (float*)   alloc((size_t)ROWS_L * 1536 * 4);
  _Float16* o2_h = (_Float16*)alloc((size_t)ROWS_L * D_MODEL * 2);
  float*    lat2 = (float*)   alloc((size_t)ROWS_L * D_MODEL * 4);
  _Float16* z_h  = (_Float16*)alloc((size_t)ROWS_L * D_MODEL * 2);
  _Float16* h_h  = (_Float16*)alloc((size_t)ROWS_L * 4 * D_MODEL * 2);

  auto cvtT = [&](const float* s, _Float16* d, int N, int K) {
    cvt_t_kernel<<<(N * K + 255) / 256, 256, 0, stream>>>(s, d, N, K);
  };
  cvtT(Wq,  WqT,  INNER, D_MODEL);            // -> [512 x 1024]
  cvtT(Wkv, WkvT, 2 * INNER, D_MODEL);        // -> [512 x 2048]
  cvtT(Wo,  WoT,  D_MODEL, INNER);            // -> [1024 x 512]
  cvtT(Wip, WipT, 3 * D_MODEL, D_MODEL);      // -> [512 x 1536]
  cvtT(Wop, WopT, D_MODEL, D_MODEL);          // -> [512 x 512]
  cvtT(Wfc, WfcT, 4 * D_MODEL, D_MODEL);      // -> [512 x 2048]
  cvtT(Wpr, WprT, D_MODEL, 4 * D_MODEL);      // -> [2048 x 512]

  // latents: LN(nl) then LN(pa_nl) -> l_h [128,512] (batch-invariant)
  ln512_double_kernel<<<NLAT / 8, 256, 0, stream>>>(lat_p, nl_g, nl_b, pl_g, pl_b, l_h);
  // visual LN -> Akv[b, 0..4095, :]
  ln512_kernel<<<(BATCH * MTOK) / 8, 256, 0, stream>>>(visual, nm_g, nm_b, Akv, 1);
  // latent rows of KV input
  bcast_lat_kernel<<<(BATCH * NLAT * D_MODEL) / 256, 256, 0, stream>>>(l_h, Akv);

  // q = (l @ Wq^T + bq) * dh^-0.5   [128, 1024]
  gemm64_kernel<<<dim3(NLAT / 64, INNER / 64), 128, 0, stream>>>(
      l_h, WqT, bq, nullptr, 0.08838834764831845f, 0, q_h, nullptr, nullptr, 0,
      INNER, D_MODEL);

  // kv = Akv @ Wkv^T + bkv  -> k^T [1024, 33792] and v [33792, 1024]
  gemm64_kernel<<<dim3(ROWS_KV / 64, 2048 / 64), 128, 0, stream>>>(
      Akv, WkvT, bkv, nullptr, 1.f, 0, kT, nullptr, vH, 2, 2048, D_MODEL);

  // flash perceiver attention -> o_h [B,128,1024]
  attn_kernel<<<dim3(BATCH * H_PA, NLAT / 16), 32, 0, stream>>>(q_h, kT, vH, o_h);

  // lat = o @ Wo^T + bo   [B*128, 512] f32
  gemm64_kernel<<<dim3(ROWS_L / 64, D_MODEL / 64), 128, 0, stream>>>(
      o_h, WoT, bo, nullptr, 1.f, 0, nullptr, lat, nullptr, 0, D_MODEL, INNER);

  // y = LN1(lat) ; qkv = y @ in_proj^T + b   [B*128, 1536] f32
  ln512_kernel<<<ROWS_L / 8, 256, 0, stream>>>(lat, ln1_g, ln1_b, y_h, 0);
  gemm64_kernel<<<dim3(ROWS_L / 64, 1536 / 64), 128, 0, stream>>>(
      y_h, WipT, bip, nullptr, 1.f, 0, nullptr, qkv, nullptr, 0, 1536, D_MODEL);

  // batch-axis MHA -> o2 f16
  mha_batch_kernel<<<NLAT, 256, 0, stream>>>(qkv, o2_h);

  // lat2 = lat + o2 @ out_proj^T + b
  gemm64_kernel<<<dim3(ROWS_L / 64, D_MODEL / 64), 128, 0, stream>>>(
      o2_h, WopT, bop, lat, 1.f, 0, nullptr, lat2, nullptr, 0, D_MODEL, D_MODEL);

  // MLP: z = LN2(lat2); h = qgelu(z @ Wfc^T + bfc); out = lat2 + h @ Wproj^T + bproj
  ln512_kernel<<<ROWS_L / 8, 256, 0, stream>>>(lat2, ln2_g, ln2_b, z_h, 0);
  gemm64_kernel<<<dim3(ROWS_L / 64, 2048 / 64), 128, 0, stream>>>(
      z_h, WfcT, bfc, nullptr, 1.f, 1, h_h, nullptr, nullptr, 0, 2048, D_MODEL);
  gemm64_kernel<<<dim3(ROWS_L / 64, D_MODEL / 64), 128, 0, stream>>>(
      h_h, WprT, bpr, lat2, 1.f, 0, nullptr, (float*)d_out, nullptr, 0, D_MODEL, 2048);
}
